// Generator_31370441130463
// MI455X (gfx1250) — matrix operand order repaired and checked by reading, not compiled, not run on hardware
//
#include <hip/hip_runtime.h>

typedef unsigned short u16;
typedef __bf16 bf16_t;
typedef bf16_t v16bf __attribute__((ext_vector_type(16)));
typedef float  v8f   __attribute__((ext_vector_type(8)));
typedef unsigned int v4u __attribute__((ext_vector_type(4)));

union FragBF { v16bf v; v4u q[2]; };

__device__ __forceinline__ u16 f2bf(float f) {
    union { float f; unsigned int u; } v; v.f = f;
    unsigned int r = v.u + 0x7FFFu + ((v.u >> 16) & 1u);  // RNE
    return (u16)(r >> 16);
}

// ---------------------------------------------------------------------------
// concat([x, emb]) -> bf16  Hc[32768][1056]
// ---------------------------------------------------------------------------
__global__ __launch_bounds__(256) void concat_cast_kernel(
    const float* __restrict__ x, const float* __restrict__ emb,
    u16* __restrict__ Hc)
{
    long long idx = (long long)blockIdx.x * 256 + threadIdx.x;
    const long long total = (long long)32768 * 1056;
    if (idx >= total) return;
    int c = (int)(idx % 1056);
    long long r = idx / 1056;
    float v;
    if (c < 1024) v = x[r * 1024 + c];
    else          v = emb[(r % 512) * 32 + (c - 1024)];
    Hc[idx] = f2bf(v);
}

// ---------------------------------------------------------------------------
// W[K][N] f32 -> Wt[N][K] bf16   (cast + transpose so GEMM B operand is [N][K])
// ---------------------------------------------------------------------------
__global__ __launch_bounds__(256) void wcast_kernel(
    const float* __restrict__ W, u16* __restrict__ Wt, int K, int N)
{
    long long idx = (long long)blockIdx.x * 256 + threadIdx.x;
    long long total = (long long)K * N;
    if (idx >= total) return;
    int n = (int)(idx % N);
    long long k = idx / N;
    Wt[(long long)n * K + k] = f2bf(W[idx]);
}

// ---------------------------------------------------------------------------
// Generic WMMA GEMM:  C[z] = act(A[z] (MxK) * Bt[z]^T (Bt is NxK) + bias)
// flags: bit0 = leaky_relu(0.2); bits[2:1] = outmode (0=f32, 1=bf16,
//        2=bf16 stored transposed per batch of sBatch rows: [b][N][sBatch])
// Double-buffered LDS, tiles filled via gfx1250 async global->LDS DMA.
// ---------------------------------------------------------------------------
#define BM 128
#define BN 128
#define BK 32
#define LDST 40   // padded LDS stride (halfwords): 40*2B = 80B -> conflict-free

__global__ __launch_bounds__(256) void wmma_gemm_kernel(
    const u16* __restrict__ A, const u16* __restrict__ Bt, void* __restrict__ Cv,
    int M, int N, int K, int lda, int ldb, int ldc,
    long long sA, long long sB, long long sC,
    const float* __restrict__ bias, int flags, int sBatch)
{
    __shared__ u16 ldsA[2][BM * LDST];
    __shared__ u16 ldsB[2][BN * LDST];

    const int tid  = threadIdx.x;
    const int lane = tid & 31;
    const int wave = tid >> 5;
    const int z    = blockIdx.z;
    A  += (long long)z * sA;
    Bt += (long long)z * sB;

    const int rowBase = blockIdx.y * BM;
    const int colBase = blockIdx.x * BN;

    const int wm = wave >> 1;   // 0..3 -> 32-row strip
    const int wn = wave & 1;    // 0..1 -> 64-col strip

    v8f acc[2][4] = {};

    const int m    = lane & 15;
    const int half = lane >> 4;

    // staging: 256 threads write only k 0-7 and 16-23 of each row of the 128x32 bf16 tile (one b128 is
    // 16B = 8 bf16) -> each thread does one 16B async op per tile, 2 threads/row
    const int ldRow = tid >> 1;          // 0..127
    const int ldCol = (tid & 1) * 16;    // 0 / 16 halfwords

    // per-thread source row pointers (B row clamped: garbage cols are never
    // stored because the epilogue guards col < N; each col depends only on
    // its own B row, so a duplicated in-bounds row is harmless)
    const u16* aRow = A + (long long)(rowBase + ldRow) * lda + ldCol;
    int nClamp = colBase + ldRow; if (nClamp > N - 1) nClamp = N - 1;
    const u16* bRow = Bt + (long long)nClamp * ldb + ldCol;

    // wave-relative LDS byte offsets for this thread's staging slots
    unsigned ldsAoff[2], ldsBoff[2];
    ldsAoff[0] = (unsigned)(unsigned long long)(uintptr_t)&ldsA[0][ldRow * LDST + ldCol];
    ldsAoff[1] = (unsigned)(unsigned long long)(uintptr_t)&ldsA[1][ldRow * LDST + ldCol];
    ldsBoff[0] = (unsigned)(unsigned long long)(uintptr_t)&ldsB[0][ldRow * LDST + ldCol];
    ldsBoff[1] = (unsigned)(unsigned long long)(uintptr_t)&ldsB[1][ldRow * LDST + ldCol];

    auto issue = [&](int k0, int buf) {
        unsigned long long ga = (unsigned long long)(uintptr_t)(aRow + k0);
        unsigned long long gb = (unsigned long long)(uintptr_t)(bRow + k0);
        asm volatile("global_load_async_to_lds_b128 %0, %1, off"
                     :: "v"(ldsAoff[buf]), "v"(ga) : "memory");
        asm volatile("global_load_async_to_lds_b128 %0, %1, off"
                     :: "v"(ldsBoff[buf]), "v"(gb) : "memory");
    };

    const int KT = K / BK;
    issue(0, 0);  // prologue: tile 0 -> buffer 0

    for (int kt = 0; kt < KT; ++kt) {
        const int cur = kt & 1;
        const int nxt = cur ^ 1;
        const int k0  = kt * BK;

        if (kt + 1 < KT) {
            issue(k0 + BK, nxt);   // fill other buffer while we compute
            asm volatile("s_wait_asynccnt 2" ::: "memory");  // tile kt done
            // deep prefetch of tile kt+2 into L2 (clamped, branch-free)
            int kp = k0 + 2 * BK; if (kp > K - BK) kp = K - BK;
            __builtin_prefetch(aRow + kp, 0, 1);
            __builtin_prefetch(bRow + kp, 0, 1);
        } else {
            asm volatile("s_wait_asynccnt 0" ::: "memory");
        }
        __syncthreads();           // all threads' tile-kt DMAs visible

        FragBF fa[2], fb[4];
        // A frag (16x32 bf16): lane holds row m, K chunks [8h..8h+7], [16+8h..]
        for (int i = 0; i < 2; ++i) {
            int r = wm * 32 + i * 16 + m;
            fa[i].q[0] = *(const v4u*)&ldsA[cur][r * LDST + 8 * half];
            fa[i].q[1] = *(const v4u*)&ldsA[cur][r * LDST + 16 + 8 * half];
        }
        // B frag (32x16 bf16, column-major source): lane holds col m, K chunks [8h..8h+7], [16+8h..]
        for (int j = 0; j < 4; ++j) {
            int c = wn * 64 + j * 16 + m;
            fb[j].q[0] = *(const v4u*)&ldsB[cur][c * LDST + 8 * half];
            fb[j].q[1] = *(const v4u*)&ldsB[cur][c * LDST + 16 + 8 * half];
        }
        for (int i = 0; i < 2; ++i)
            for (int j = 0; j < 4; ++j)
                acc[i][j] = __builtin_amdgcn_wmma_f32_16x16x32_bf16(
                    false, fa[i].v, false, fb[j].v,
                    (short)0, acc[i][j], false, false);

        __syncthreads();           // done reading `cur` before it refills
    }

    // epilogue; C 16x16 layout: elem e of v8f = C[e + 8*half][m]
    const bool act   = (flags & 1) != 0;
    const int  omode = (flags >> 1) & 3;
    for (int i = 0; i < 2; ++i)
        for (int j = 0; j < 4; ++j) {
            int col = colBase + wn * 64 + j * 16 + m;
            if (col >= N) continue;
            float bv = bias ? bias[col] : 0.0f;
            int rbase = rowBase + wm * 32 + i * 16 + 8 * half;
            for (int e = 0; e < 8; ++e) {
                float v = acc[i][j][e] + bv;
                if (act) v = (v > 0.0f) ? v : 0.2f * v;
                int row = rbase + e;
                if (omode == 0) {
                    ((float*)Cv)[(long long)z * sC + (long long)row * ldc + col] = v;
                } else if (omode == 1) {
                    ((u16*)Cv)[(long long)z * sC + (long long)row * ldc + col] = f2bf(v);
                } else {
                    int b = row / sBatch, s = row % sBatch;
                    ((u16*)Cv)[((long long)b * N + col) * sBatch + s] = f2bf(v);
                }
            }
        }
}

// ---------------------------------------------------------------------------
// row softmax: E[32768][512] f32 -> P[32768][512] bf16
// ---------------------------------------------------------------------------
__global__ __launch_bounds__(256) void softmax_kernel(
    const float* __restrict__ E, u16* __restrict__ P)
{
    __shared__ float red[256];
    const int row = blockIdx.x;
    const float* e = E + (long long)row * 512;
    const int t = threadIdx.x;

    float a = e[t], b = e[t + 256];
    red[t] = fmaxf(a, b);
    __syncthreads();
    for (int s = 128; s > 0; s >>= 1) {
        if (t < s) red[t] = fmaxf(red[t], red[t + s]);
        __syncthreads();
    }
    float mx = red[0];
    __syncthreads();

    float ea = __expf(a - mx), eb = __expf(b - mx);
    red[t] = ea + eb;
    __syncthreads();
    for (int s = 128; s > 0; s >>= 1) {
        if (t < s) red[t] += red[t + s];
        __syncthreads();
    }
    float inv = 1.0f / red[0];

    u16* p = P + (long long)row * 512;
    p[t]       = f2bf(ea * inv);
    p[t + 256] = f2bf(eb * inv);
}

// ---------------------------------------------------------------------------
extern "C" void kernel_launch(void* const* d_in, const int* in_sizes, int n_in,
                              void* d_out, int out_size, void* d_ws, size_t ws_size,
                              hipStream_t stream)
{
    (void)in_sizes; (void)n_in; (void)out_size; (void)ws_size;

    const float* x   = (const float*)d_in[0];
    const float* emb = (const float*)d_in[1];
    const float* w1  = (const float*)d_in[2];
    const float* b1  = (const float*)d_in[3];
    const float* wq  = (const float*)d_in[4];
    const float* bq  = (const float*)d_in[5];
    const float* wk  = (const float*)d_in[6];
    const float* bk  = (const float*)d_in[7];
    const float* wv  = (const float*)d_in[8];
    const float* bv  = (const float*)d_in[9];
    const float* w2  = (const float*)d_in[10];
    const float* b2  = (const float*)d_in[11];
    float* out = (float*)d_out;

    constexpr int SQ = 512, BAT = 64, DM = 1024, IN = 1056;
    constexpr int MROWS = BAT * SQ;                       // 32768
    constexpr size_t wtsz  = (size_t)IN * IN * 2;         // bf16 [N][K]
    constexpr size_t wt2sz = (size_t)DM * IN * 2;
    constexpr size_t bigsz = (size_t)MROWS * IN * 2;      // 69.2 MB

    size_t off = 0;
    auto alloc = [&](size_t sz) -> void* {
        void* p = (char*)d_ws + off;
        off += (sz + 255) & ~(size_t)255;
        return p;
    };
    u16* Wt1  = (u16*)alloc(wtsz);
    u16* WtQ  = (u16*)alloc(wtsz);
    u16* WtK  = (u16*)alloc(wtsz);
    u16* WtV  = (u16*)alloc(wtsz);
    u16* Wt2  = (u16*)alloc(wt2sz);
    u16* Abuf = (u16*)alloc(bigsz);  // Hcat -> Q -> O
    u16* Bbuf = (u16*)alloc(bigsz);  // H -> E (f32, 67MB fits)
    u16* Kbuf = (u16*)alloc(bigsz);  // K
    u16* Vtb  = (u16*)alloc(bigsz);  // V^T per batch: [b][IN][SQ]
    u16* Pbuf = (u16*)alloc(bigsz);  // softmax(P) bf16 (33.5MB used)

    // 1) concat + cast
    {
        long long total = (long long)MROWS * IN;
        int blocks = (int)((total + 255) / 256);
        concat_cast_kernel<<<blocks, 256, 0, stream>>>(x, emb, Abuf);
    }
    // 2) weight cast + transpose
    {
        int blk = (IN * IN + 255) / 256;
        wcast_kernel<<<blk, 256, 0, stream>>>(w1, Wt1, IN, IN);
        wcast_kernel<<<blk, 256, 0, stream>>>(wq, WtQ, IN, IN);
        wcast_kernel<<<blk, 256, 0, stream>>>(wk, WtK, IN, IN);
        wcast_kernel<<<blk, 256, 0, stream>>>(wv, WtV, IN, IN);
        int blk2 = (IN * DM + 255) / 256;
        wcast_kernel<<<blk2, 256, 0, stream>>>(w2, Wt2, IN, DM);
    }
    // 3) H = leaky_relu(Hc*W1 + b1)  -> bf16
    {
        dim3 g((IN + BN - 1) / BN, MROWS / BM, 1);
        wmma_gemm_kernel<<<g, 256, 0, stream>>>(Abuf, Wt1, Bbuf,
            MROWS, IN, IN, IN, IN, IN, 0, 0, 0, b1, /*act|bf16*/ 1 | (1 << 1), SQ);
    }
    // 4) Q, K projections -> bf16 ; V -> V^T per batch
    {
        dim3 g((IN + BN - 1) / BN, MROWS / BM, 1);
        wmma_gemm_kernel<<<g, 256, 0, stream>>>(Bbuf, WtQ, Abuf,
            MROWS, IN, IN, IN, IN, IN, 0, 0, 0, bq, (1 << 1), SQ);
        wmma_gemm_kernel<<<g, 256, 0, stream>>>(Bbuf, WtK, Kbuf,
            MROWS, IN, IN, IN, IN, IN, 0, 0, 0, bk, (1 << 1), SQ);
        wmma_gemm_kernel<<<g, 256, 0, stream>>>(Bbuf, WtV, Vtb,
            MROWS, IN, IN, IN, IN, IN, 0, 0, 0, bv, (2 << 1), SQ);
    }
    // 5) energy = Q*K^T per batch -> f32 (K rows are already the [N][K] operand)
    {
        dim3 g(SQ / BN, SQ / BM, BAT);
        wmma_gemm_kernel<<<g, 256, 0, stream>>>(Abuf, Kbuf, Bbuf,
            SQ, SQ, IN, IN, IN, SQ,
            (long long)SQ * IN, (long long)SQ * IN, (long long)SQ * SQ,
            nullptr, 0, SQ);
    }
    // 6) softmax rows -> bf16 P
    softmax_kernel<<<MROWS, 256, 0, stream>>>((const float*)Bbuf, Pbuf);
    // 7) O = P * V per batch (B operand = V^T, already [N=IN][K=SQ]) -> bf16
    {
        dim3 g((IN + BN - 1) / BN, SQ / BM, BAT);
        wmma_gemm_kernel<<<g, 256, 0, stream>>>(Pbuf, Vtb, Abuf,
            SQ, IN, SQ, SQ, SQ, IN,
            (long long)SQ * SQ, (long long)IN * SQ, (long long)SQ * IN,
            nullptr, (1 << 1), SQ);
    }
    // 8) out = O * W2 + b2 -> f32
    {
        dim3 g(DM / BN, MROWS / BM, 1);
        wmma_gemm_kernel<<<g, 256, 0, stream>>>(Abuf, Wt2, out,
            MROWS, DM, IN, IN, IN, DM, 0, 0, 0, b2, 0, SQ);
    }
}
